// SpatialBlock_67611375173666
// MI455X (gfx1250) — compile-verified, matrix operand order checked by reading, not run on hardware
//
#include <hip/hip_runtime.h>
#include <hip/hip_bf16.h>

// Problem sizes (fixed by the reference).
#define B_ROWS 16384
#define NN     8192
#define DIN    256
#define DOUT   512

typedef __attribute__((ext_vector_type(16))) __bf16        v16bf;
typedef __attribute__((ext_vector_type(8)))  float         v8f;
typedef __attribute__((ext_vector_type(4)))  unsigned int  v4u;
typedef __attribute__((ext_vector_type(8)))  int           v8i;
typedef __attribute__((ext_vector_type(4)))  int           v4i;

union Frag16 {
    uint4 u4[2];
    v16bf v;
};

__device__ __forceinline__ v8f vzero8() {
    v8f z = {0.f, 0.f, 0.f, 0.f, 0.f, 0.f, 0.f, 0.f};
    return z;
}

// ---------------------------------------------------------------------------
// LDS layout (dynamic shared, one block per WGP; ~252.5 KB of the 320 KB WGP
// budget).  Rows are padded by 16B (TDM pad_enable) so the per-lane 16-byte
// fragment strips hit 16 different banks instead of 2.
// ---------------------------------------------------------------------------
#define XROWB     528                         // 256 bf16 (512B) + 16B pad
#define XBUF_OFF  0
#define XBUF_SZ   (64 * XROWB)                // 33792
#define PBUF_OFF  XBUF_SZ                     // 33792
#define PBUF_SZ   (64 * XROWB)                // 33792 (x2 buffers)
#define WROWB     144                         // 64 bf16 (128B) + 16B pad
#define WBUF_OFF  (PBUF_OFF + 2 * PBUF_SZ)    // 101376
#define WBUF_SZ   (512 * WROWB)               // 73728 (x2 buffers)
#define GWROWB    144                         // 64 bf16 + 16B pad
#define GW_OFF    (WBUF_OFF + 2 * WBUF_SZ)    // 248832
#define GW_SZ     (64 * GWROWB)               // 9216
#define X2_OFF    (GW_OFF + GW_SZ)            // 258048
#define RS_OFF    (X2_OFF + 256)              // 258304
#define SMEM_BYTES (RS_OFF + 256)             // 258560

// ---------------------------------------------------------------------------
// Tensor Data Mover: issue a 2-D tile load (global -> LDS) per ISA 08 D# spec.
// Group0: count=1 | lds_addr | global_addr | type=2.
// Group1: data_size=2B, pad_enable, pad_interval/amount, tensor/tile dims,
//         dim0 stride.  Remaining groups unused (2-D tensor) -> zero.
// This toolchain's builtin takes 6 args:
//   (uint32x4 g0, int32x8 g1, int32x4, int32x4, int32x8, i32 cpol)
// Tracked by TENSORcnt; EXEC-independent, issue from one wave only.
// ---------------------------------------------------------------------------
__device__ __forceinline__ void tdm_load_2d(unsigned int lds_off, const void* gptr,
                                            unsigned int tensor_d0, unsigned int tensor_d1,
                                            unsigned int tile_d0,   unsigned int tile_d1,
                                            unsigned int stride0,
                                            unsigned int pad_interval_code,
                                            unsigned int pad_amount_code) {
    const unsigned long long ga = (unsigned long long)(uintptr_t)gptr;
    v4u g0;
    g0[0] = 1u;                                        // count=1, user desc
    g0[1] = lds_off;                                   // lds_addr (bytes)
    g0[2] = (unsigned int)ga;                          // global_addr[31:0]
    g0[3] = (unsigned int)(ga >> 32) | (2u << 30);     // addr[56:32] | type=2
    v8i g1;
    g1[0] = (int)((1u << 16) |                         // data_size = 2 bytes
                  (1u << 20) |                         // pad_enable
                  (pad_interval_code << 22) |
                  (pad_amount_code << 25));
    g1[1] = (int)(tensor_d0 << 16);                    // tensor_dim0[15:0]
    g1[2] = (int)((tensor_d0 >> 16) | (tensor_d1 << 16));
    g1[3] = (int)((tensor_d1 >> 16) | (tile_d0 << 16));
    g1[4] = (int)(tile_d1 & 0xffffu);                  // tile_dim1 (tile_dim2=0)
    g1[5] = (int)stride0;                              // dim0 stride[31:0]
    g1[6] = 0;
    g1[7] = 0;
    const v4i z4 = {0, 0, 0, 0};
    const v8i z8 = {0, 0, 0, 0, 0, 0, 0, 0};
    __builtin_amdgcn_tensor_load_to_lds(g0, g1, z4, z4, z8, 0);
}

// ---------------------------------------------------------------------------
// 16x32 bf16 A/B-style WMMA fragment from an LDS tile with padded row stride.
// CDNA5 16-bit fragment layout (ISA 7.12.2):
//   lanes 0-15 : row=row0+lane,    VGPR0-3 = K k0..+7,   VGPR4-7 = K k0+16..+23
//   lanes 16-31: row=row0+lane-16, VGPR0-3 = K k0+8..+15, VGPR4-7 = K k0+24..+31
// row_stride_bytes and k0*2 are multiples of 16 -> two ds_load_b128 per lane.
// ---------------------------------------------------------------------------
__device__ __forceinline__ v16bf load_frag_lds(const char* base, int row0,
                                               int row_stride_bytes, int k0, int lane) {
    const int r  = row0 + (lane & 15);
    const int kb = (k0 + ((lane >> 4) << 3)) * 2;
    const char* p = base + r * row_stride_bytes + kb;
    Frag16 f;
    f.u4[0] = *(const uint4*)(p);        // elements k .. k+7
    f.u4[1] = *(const uint4*)(p + 32);   // elements k+16 .. k+23
    return f.v;
}

// ---------------------------------------------------------------------------
// Pre-pass 1: x -> bf16 copy + row sumsq  (one 256-thread block per row)
// ---------------------------------------------------------------------------
__global__ void xprep_kernel(const float* __restrict__ x,
                             __hip_bfloat16* __restrict__ xbf,
                             float* __restrict__ x2) {
    __shared__ float red[256];
    const int row = blockIdx.x;
    const int t   = threadIdx.x;
    const float v = x[(size_t)row * DIN + t];
    xbf[(size_t)row * DIN + t] = __float2bfloat16(v);
    red[t] = v * v;
    __syncthreads();
    for (int s = 128; s > 0; s >>= 1) {
        if (t < s) red[t] += red[t + s];
        __syncthreads();
    }
    if (t == 0) x2[row] = red[0];
}

// ---------------------------------------------------------------------------
// Pre-pass 2: positions -> bf16 + ||p||^2 + cc = log2e / (2*sigma^2)
// ---------------------------------------------------------------------------
__global__ void pprep_kernel(const float* __restrict__ p,
                             const float* __restrict__ ls,
                             __hip_bfloat16* __restrict__ pbf,
                             float* __restrict__ p2,
                             float* __restrict__ cc) {
    __shared__ float red[256];
    const int row = blockIdx.x;
    const int t   = threadIdx.x;
    const float v = p[(size_t)row * DIN + t];
    pbf[(size_t)row * DIN + t] = __float2bfloat16(v);
    red[t] = v * v;
    __syncthreads();
    for (int s = 128; s > 0; s >>= 1) {
        if (t < s) red[t] += red[t + s];
        __syncthreads();
    }
    if (t == 0) {
        p2[row] = red[0];
        const float l  = ls[row];
        const float sp = (l > 20.f) ? l : log1pf(expf(l));   // softplus
        const float sg = sp + 1e-6f;
        cc[row] = 1.4426950408889634f / (2.f * sg * sg);     // log2(e)/(2s^2)
    }
}

// ---------------------------------------------------------------------------
// Pre-pass 3: W (N x DOUT, f32) -> W^T (DOUT x N, bf16) so GEMM2 B-fragments
// are contiguous K-strips and the TDM tile is a simple strided 2-D tile.
// ---------------------------------------------------------------------------
__global__ void wtrans_kernel(const float* __restrict__ w,
                              __hip_bfloat16* __restrict__ wt) {
    __shared__ float tile[64][65];
    const int d0 = blockIdx.x * 64;
    const int n0 = blockIdx.y * 64;
    const int tx = threadIdx.x & 63;
    const int ty = threadIdx.x >> 6;        // 0..3
#pragma unroll
    for (int r = 0; r < 16; ++r) {
        const int nr = ty + r * 4;
        tile[nr][tx] = w[(size_t)(n0 + nr) * DOUT + d0 + tx];
    }
    __syncthreads();
#pragma unroll
    for (int r = 0; r < 16; ++r) {
        const int dr = ty + r * 4;
        wt[(size_t)(d0 + dr) * NN + n0 + tx] = __float2bfloat16(tile[tx][dr]);
    }
}

// ---------------------------------------------------------------------------
// Fused main kernel (64 rows/block, 16 wave32s).  Per 64-neuron tile:
//   TDM prefetch of next P/W^T tiles into the idle LDS buffers (wave 0),
//   GEMM1 (bf16 WMMA) from LDS, gw = exp2(-max(d2,0)*cc) + rowsums,
//   GEMM2 (bf16 WMMA) accumulating O(64x512) in registers,
//   s_wait_tensorcnt + barrier to flip buffers.  Epilogue normalizes.
// ---------------------------------------------------------------------------
__global__ __launch_bounds__(512)
void spatial_main(const __hip_bfloat16* __restrict__ xbf,
                  const __hip_bfloat16* __restrict__ pbf,
                  const __hip_bfloat16* __restrict__ wtbf,
                  const float* __restrict__ x2g,
                  const float* __restrict__ p2g,
                  const float* __restrict__ ccg,
                  float* __restrict__ out) {
    extern __shared__ __align__(16) char smem[];
    __hip_bfloat16* lds_gw = (__hip_bfloat16*)(smem + GW_OFF);
    float* s_x2   = (float*)(smem + X2_OFF);
    float* s_rsum = (float*)(smem + RS_OFF);

    const int tid  = threadIdx.x;
    const int lane = tid & 31;
    const int wid  = tid >> 5;          // 0..15 (wave32)
    const int row0 = blockIdx.x * 64;
    const unsigned int lds_base = (unsigned int)(uintptr_t)smem;

    if (tid < 64) {
        s_x2[tid]   = x2g[row0 + tid];
        s_rsum[tid] = 0.f;
    }

    // Prologue: DMA X block (once) + first P / W^T tiles into buffer 0.
    if (wid == 0) {
        tdm_load_2d(lds_base + XBUF_OFF, xbf + (size_t)row0 * DIN,
                    DIN, B_ROWS, DIN, 64, DIN, /*pad: 512B rows*/ 6, 3);
        tdm_load_2d(lds_base + PBUF_OFF, pbf,
                    DIN, NN, DIN, 64, DIN, 6, 3);
        tdm_load_2d(lds_base + WBUF_OFF, wtbf,
                    NN, DOUT, 64, DOUT, NN, /*pad: 128B rows*/ 4, 3);
        __builtin_amdgcn_s_wait_tensorcnt(0);
    }
    __syncthreads();

    const int mi    = wid & 3;          // 16x16 tile row (both GEMMs)
    const int ni    = wid >> 2;         // GEMM1 tile col
    const int njb   = (wid >> 2) * 8;   // GEMM2 first output n-tile
    const int nhalf = lane >> 4;        // C layout: rows r (0) / r+8 (1)
    const int nlo   = lane & 15;        // column within 16x16 tile

    v8f acc[8];
#pragma unroll
    for (int t = 0; t < 8; ++t) acc[t] = vzero8();

    int cur = 0;
    for (int n0 = 0; n0 < NN; n0 += 64) {
        const int nxt = cur ^ 1;
        // Kick off next tile's DMA immediately; it overlaps both GEMMs.
        if (wid == 0 && n0 + 64 < NN) {
            tdm_load_2d(lds_base + PBUF_OFF + nxt * PBUF_SZ,
                        pbf + (size_t)(n0 + 64) * DIN,
                        DIN, NN, DIN, 64, DIN, 6, 3);
            tdm_load_2d(lds_base + WBUF_OFF + nxt * WBUF_SZ,
                        wtbf + (n0 + 64),
                        NN, DOUT, 64, DOUT, NN, 4, 3);
        }

        // ---- GEMM1: one 16x16 S-tile per wave, K = 256 = 8 x 32 ----
        const char* xb = smem + XBUF_OFF;
        const char* pb = smem + PBUF_OFF + cur * PBUF_SZ;
        v8f s = vzero8();
#pragma unroll
        for (int kt = 0; kt < 8; ++kt) {
            v16bf a = load_frag_lds(xb, mi * 16, XROWB, kt * 32, lane);
            v16bf b = load_frag_lds(pb, ni * 16, XROWB, kt * 32, lane);
            s = __builtin_amdgcn_wmma_f32_16x16x32_bf16(false, a, false, b,
                                                        (short)0, s, false, false);
        }

        // ---- gaussian weights + row-sum partials ----
        const int   ncol = n0 + ni * 16 + nlo;
        const float p2v  = p2g[ncol];
        const float ccv  = ccg[ncol];
        float rs[8];
#pragma unroll
        for (int r = 0; r < 8; ++r) {
            const int   ml = mi * 16 + r + 8 * nhalf;       // local row 0..63
            const float d2 = fmaxf(s_x2[ml] + p2v - 2.f * s[r], 0.f);
            const float g  = exp2f(-d2 * ccv);
            rs[r] = g;
            lds_gw[ml * (GWROWB / 2) + ni * 16 + nlo] = __float2bfloat16(g);
        }
        // reduce each row's 16 columns across lanes {0..15} / {16..31}
#pragma unroll
        for (int r = 0; r < 8; ++r) {
            float v = rs[r];
            v += __shfl_xor(v, 1, 32);
            v += __shfl_xor(v, 2, 32);
            v += __shfl_xor(v, 4, 32);
            v += __shfl_xor(v, 8, 32);
            rs[r] = v;
        }
        if (nlo == 0) {                    // lanes 0 and 16
#pragma unroll
            for (int r = 0; r < 8; ++r)
                atomicAdd(&s_rsum[mi * 16 + r + 8 * nhalf], rs[r]);
        }
        __syncthreads();                   // publish gw

        // ---- GEMM2: acc[t] += gw(16x64) @ W_tile(64x16), K = 2 x 32 ----
        const char* wb = smem + WBUF_OFF + cur * WBUF_SZ;
#pragma unroll
        for (int s2 = 0; s2 < 2; ++s2) {
            v16bf a2 = load_frag_lds(smem + GW_OFF, mi * 16, GWROWB, s2 * 32, lane);
#pragma unroll
            for (int t = 0; t < 8; ++t) {
                v16bf b2 = load_frag_lds(wb, (njb + t) * 16, WROWB, s2 * 32, lane);
                acc[t] = __builtin_amdgcn_wmma_f32_16x16x32_bf16(false, a2, false, b2,
                                                                 (short)0, acc[t],
                                                                 false, false);
            }
        }

        if (wid == 0) __builtin_amdgcn_s_wait_tensorcnt(0);  // next buffers landed
        __syncthreads();                   // flip buffers / free gw
        cur = nxt;
    }

    // ---- epilogue: normalize by row sums and store ----
#pragma unroll
    for (int t = 0; t < 8; ++t) {
        const int col = (njb + t) * 16 + nlo;
#pragma unroll
        for (int r = 0; r < 8; ++r) {
            const int   ml  = mi * 16 + r + 8 * nhalf;
            const float inv = 1.f / (s_rsum[ml] + 1e-8f);
            out[(size_t)(row0 + ml) * DOUT + col] = acc[t][r] * inv;
        }
    }
}

// ---------------------------------------------------------------------------
// Host-side launcher. Workspace layout (bytes, all 16B-aligned), ~20.2 MB:
//   [0)          xbf  : B*DIN  bf16  = 8388608
//   [8388608)    pbf  : N*DIN  bf16  = 4194304
//   [12582912)   wtbf : DOUT*N bf16  = 8388608
//   [20971520)   x2   : B      f32   = 65536
//   [21037056)   p2   : N      f32   = 32768
//   [21069824)   cc   : N      f32   = 32768
// ---------------------------------------------------------------------------
extern "C" void kernel_launch(void* const* d_in, const int* in_sizes, int n_in,
                              void* d_out, int out_size, void* d_ws, size_t ws_size,
                              hipStream_t stream) {
    const float* x   = (const float*)d_in[0];
    const float* pos = (const float*)d_in[1];
    const float* w   = (const float*)d_in[2];
    const float* lsg = (const float*)d_in[3];
    float* out = (float*)d_out;

    char* ws = (char*)d_ws;
    __hip_bfloat16* xbf  = (__hip_bfloat16*)(ws);
    __hip_bfloat16* pbf  = (__hip_bfloat16*)(ws + 8388608);
    __hip_bfloat16* wtbf = (__hip_bfloat16*)(ws + 12582912);
    float* x2 = (float*)(ws + 20971520);
    float* p2 = (float*)(ws + 21037056);
    float* cc = (float*)(ws + 21069824);

    xprep_kernel<<<B_ROWS, 256, 0, stream>>>(x, xbf, x2);
    pprep_kernel<<<NN, 256, 0, stream>>>(pos, lsg, pbf, p2, cc);
    wtrans_kernel<<<dim3(DOUT / 64, NN / 64), 256, 0, stream>>>(w, wtbf);
    spatial_main<<<B_ROWS / 64, 512, SMEM_BYTES, stream>>>(xbf, pbf, wtbf,
                                                           x2, p2, cc, out);
}
